// DUST_V2_65085934403761
// MI455X (gfx1250) — matrix operand order repaired and checked by reading, not compiled, not run on hardware
//
#include <hip/hip_runtime.h>
#include <math.h>

// ---------------- problem constants (match reference) ----------------
#define D_DIM   1024
#define W_HALF  512
#define BATCH   4096
#define P_PREV  8
#define OMEGA   10
#define N_IT    10

// ---------------- workspace layout (float offsets) ----------------
#define OFS_B      0u          // [4096*1024] f32  b = x@Wd.T / L
#define OFS_STR    4194304u    // [1024*1024] f32  ST = S0^T (sparse steps)
#define OFS_AH     5242880u    // [4096*1024] bf16 hi of x / z_att
#define OFS_AL     7340032u    // [4096*1024] bf16 lo of x / z_att
#define OFS_WH     9437184u    // [1024*1024] bf16 hi of Wd
#define OFS_WL     9961472u    // [1024*1024] bf16 lo of Wd
#define OFS_SH     10485760u   // [1024*1024] bf16 hi of S
#define OFS_SL     11010048u   // [1024*1024] bf16 lo of S
#define OFS_ZVAL   11534336u   // [4096*10]  sparse z values
#define OFS_ZIDX   11575296u   // [4096*10]  sparse z indices (int)
#define OFS_DENS0  11616256u   // [512] density accumulator (warm-up z)
#define OFS_DENS1  11616768u   // [512] density accumulator (final z)
#define OFS_MDPREV 11617280u   // [8*512] prev-window densities
#define OFS_MDN    11621376u   // [512] normalized density of warm-up z
#define OFS_ATT    11621888u   // [8] attention weights

typedef __attribute__((ext_vector_type(16))) __bf16 v16bf;
typedef __attribute__((ext_vector_type(8)))  __bf16 v8bf;
typedef __attribute__((ext_vector_type(4)))  __bf16 v4bf;
typedef __attribute__((ext_vector_type(8)))  float  v8f;
typedef __attribute__((ext_vector_type(4)))  int    v4i;

// ---- CDNA5 async global->LDS copy (guarded; falls back to sync copy) ----
#if defined(__has_builtin)
#if __has_builtin(__builtin_amdgcn_global_load_async_to_lds_b128) && \
    __has_builtin(__builtin_amdgcn_s_wait_asynccnt)
#define USE_ASYNC_LDS 1
#endif
#endif
#ifndef USE_ASYNC_LDS
#define USE_ASYNC_LDS 0
#endif

#if USE_ASYNC_LDS
typedef v4i __attribute__((address_space(1))) *v4i_gptr;   // global int4*
typedef v4i __attribute__((address_space(3))) *v4i_lptr;   // LDS int4*
#endif

__device__ __forceinline__ void copy16_g2l(const __bf16* g, __bf16* l)
{
#if USE_ASYNC_LDS
  __builtin_amdgcn_global_load_async_to_lds_b128(
      (v4i_gptr)(g), (v4i_lptr)(l), 0, 0);
#else
  *(uint4*)l = *(const uint4*)g;
#endif
}

__device__ __forceinline__ void async_stage_fence()
{
#if USE_ASYNC_LDS
  __builtin_amdgcn_s_wait_asynccnt(0);
#endif
  __syncthreads();
}

// concat two 16-byte LDS chunks into one v16bf fragment register set
__device__ __forceinline__ v16bf frag_cat(const __bf16* p0, const __bf16* p1)
{
  v8bf a = *(const v8bf*)p0;
  v8bf b = *(const v8bf*)p1;
  return __builtin_shufflevector(a, b, 0, 1, 2, 3, 4, 5, 6, 7,
                                 8, 9, 10, 11, 12, 13, 14, 15);
}

// =====================================================================
// WMMA GEMM (NT) on pre-packed hi/lo bf16:
//   C[m][n] = alpha * sum_k (Ah+Al)[m][k] * (Bh+Bl)[n][k] (+ addsrc)
// bf16x3: hi*hi + hi*lo + lo*hi, f32 accumulate.
// =====================================================================
#define BM 128
#define BN 64
#define BK 32
#define LDSPB 40   // bf16 per LDS row: 80B stride, 16B aligned, conflict-free

__global__ __launch_bounds__(256)
void gemm_nt_bf16pk(const __bf16* __restrict__ Ah, const __bf16* __restrict__ Al,
                    const __bf16* __restrict__ Bh, const __bf16* __restrict__ Bl,
                    float* __restrict__ C, const float* __restrict__ addsrc,
                    const float* __restrict__ inv_scale_src,
                    int M, int N, int K)
{
  __shared__ __bf16 lsAh[BM * LDSPB];   // 10 KB
  __shared__ __bf16 lsAl[BM * LDSPB];   // 10 KB
  __shared__ __bf16 lsBh[BN * LDSPB];   //  5 KB
  __shared__ __bf16 lsBl[BN * LDSPB];   //  5 KB
  const int tid  = threadIdx.x;
  const int lane = tid & 31;
  const int wid  = tid >> 5;
  const int wm   = wid & 3;          // 4 waves along M
  const int wn   = wid >> 2;         // 2 waves along N
  const int m0   = blockIdx.y * BM;
  const int n0   = blockIdx.x * BN;

  v8f acc[2][2] = {};

  for (int kk = 0; kk < K; kk += BK) {
    // A tiles: 128x32 bf16 = 512 x 16B each (hi+lo), 2 per thread per tile
#pragma unroll
    for (int i = 0; i < 2; ++i) {
      int f = i * 256 + tid;           // 0..511
      int r = f >> 2, c = f & 3;       // 4 x 16B per row
      const size_t go = (size_t)(m0 + r) * K + kk + c * 8;
      copy16_g2l(Ah + go, &lsAh[r * LDSPB + c * 8]);
      copy16_g2l(Al + go, &lsAl[r * LDSPB + c * 8]);
    }
    // B tiles: 64x32 bf16 = 256 x 16B each (hi+lo), 1 per thread per tile
    {
      int r = tid >> 2, c = tid & 3;
      const size_t go = (size_t)(n0 + r) * K + kk + c * 8;
      copy16_g2l(Bh + go, &lsBh[r * LDSPB + c * 8]);
      copy16_g2l(Bl + go, &lsBl[r * LDSPB + c * 8]);
    }
    async_stage_fence();

    // A frag: lane -> row (lane&15); elems 0..7 = K k0..k0+7,
    // elems 8..15 = K k0+16..k0+23, k0 = (lane>>4)*8.
    v16bf ah[2], al[2], bh[2], bl[2];
#pragma unroll
    for (int i = 0; i < 2; ++i) {
      int row = wm * 32 + i * 16 + (lane & 15);
      int k0  = (lane >> 4) << 3;
      ah[i] = frag_cat(&lsAh[row * LDSPB + k0], &lsAh[row * LDSPB + k0 + 16]);
      al[i] = frag_cat(&lsAl[row * LDSPB + k0], &lsAl[row * LDSPB + k0 + 16]);
    }
    // B frag: lane -> col (lane&15); elems 0..7 = K kb..kb+7,
    // elems 8..15 = K kb+16..kb+23, kb = (lane>>4)*8.
#pragma unroll
    for (int j = 0; j < 2; ++j) {
      int row = wn * 32 + j * 16 + (lane & 15);
      int kb  = (lane >> 4) << 3;
      bh[j] = frag_cat(&lsBh[row * LDSPB + kb], &lsBh[row * LDSPB + kb + 16]);
      bl[j] = frag_cat(&lsBl[row * LDSPB + kb], &lsBl[row * LDSPB + kb + 16]);
    }

#pragma unroll
    for (int i = 0; i < 2; ++i)
#pragma unroll
      for (int j = 0; j < 2; ++j) {
        acc[i][j] = __builtin_amdgcn_wmma_f32_16x16x32_bf16(
            false, ah[i], false, bh[j], (short)0, acc[i][j], false, false);
        acc[i][j] = __builtin_amdgcn_wmma_f32_16x16x32_bf16(
            false, ah[i], false, bl[j], (short)0, acc[i][j], false, false);
        acc[i][j] = __builtin_amdgcn_wmma_f32_16x16x32_bf16(
            false, al[i], false, bh[j], (short)0, acc[i][j], false, false);
      }
    __syncthreads();
  }

  const float alpha = inv_scale_src ? (1.0f / inv_scale_src[0]) : 1.0f;
  const int mh = (lane >> 4) << 3;   // C layout: lanes 16..31 hold M=r+8
  const int nl = lane & 15;
#pragma unroll
  for (int i = 0; i < 2; ++i)
#pragma unroll
    for (int j = 0; j < 2; ++j)
#pragma unroll
      for (int r = 0; r < 8; ++r) {
        int m = m0 + wm * 32 + i * 16 + r + mh;
        int n = n0 + wn * 32 + j * 16 + nl;
        float v = acc[i][j][r] * alpha;
        if (addsrc) v += addsrc[(size_t)m * N + n];
        C[(size_t)m * N + n] = v;
      }
}

// split f32 -> bf16 hi + bf16 lo (one pass over the matrix, float4 granular)
__global__ __launch_bounds__(256)
void prepack_bf16x2(const float* __restrict__ src, __bf16* __restrict__ hi,
                    __bf16* __restrict__ lo, int n4)
{
  int g = blockIdx.x * blockDim.x + threadIdx.x;
  if (g >= n4) return;
  float4 v = ((const float4*)src)[g];
  v4bf h, l;
  __bf16 t;
  t = (__bf16)v.x; h[0] = t; l[0] = (__bf16)(v.x - (float)t);
  t = (__bf16)v.y; h[1] = t; l[1] = (__bf16)(v.y - (float)t);
  t = (__bf16)v.z; h[2] = t; l[2] = (__bf16)(v.z - (float)t);
  t = (__bf16)v.w; h[3] = t; l[3] = (__bf16)(v.w - (float)t);
  *(v4bf*)(hi + 4 * (size_t)g) = h;
  *(v4bf*)(lo + 4 * (size_t)g) = l;
}

// =====================================================================
// top-OMEGA |value| selection within one wave32 row (32 f32 per lane)
// element j = (u*32+lane)*4 + s ; writes sparse (val, idx) list
// =====================================================================
__device__ __forceinline__ void topk_select(float4 (&v)[8], int lane, int row,
                                            float* __restrict__ zval,
                                            int*   __restrict__ zidx,
                                            unsigned &taken)
{
  taken = 0u;
#pragma unroll 1
  for (int t = 0; t < OMEGA; ++t) {
    float bv = -1.0f, bs = 0.0f; int bj = 1 << 30;
#pragma unroll
    for (int u = 0; u < 8; ++u) {
      const int jb = (u * 32 + lane) << 2;
      float a;
      a = fabsf(v[u].x); if (!((taken >> (u*4+0)) & 1u) && a > bv) { bv = a; bs = v[u].x; bj = jb + 0; }
      a = fabsf(v[u].y); if (!((taken >> (u*4+1)) & 1u) && a > bv) { bv = a; bs = v[u].y; bj = jb + 1; }
      a = fabsf(v[u].z); if (!((taken >> (u*4+2)) & 1u) && a > bv) { bv = a; bs = v[u].z; bj = jb + 2; }
      a = fabsf(v[u].w); if (!((taken >> (u*4+3)) & 1u) && a > bv) { bv = a; bs = v[u].w; bj = jb + 3; }
    }
#pragma unroll
    for (int off = 16; off > 0; off >>= 1) {
      float ov = __shfl_xor(bv, off, 32);
      float os = __shfl_xor(bs, off, 32);
      int   oj = __shfl_xor(bj, off, 32);
      if (ov > bv || (ov == bv && oj < bj)) { bv = ov; bs = os; bj = oj; }
    }
    const int owner = (bj >> 2) & 31;
    if (lane == owner) taken |= 1u << (((bj >> 7) << 2) | (bj & 3));
    if (lane == 0) { zval[row * OMEGA + t] = bs; zidx[row * OMEGA + t] = bj; }
  }
}

// hard_thr on a dense [BATCH, D] matrix -> sparse list. 1 wave per row.
__global__ __launch_bounds__(256)
void hardthr_rows(const float* __restrict__ src,
                  float* __restrict__ zval, int* __restrict__ zidx)
{
  const int lane = threadIdx.x & 31;
  const int row  = blockIdx.x * 8 + (threadIdx.x >> 5);
  const float4* s4 = (const float4*)(src + (size_t)row * D_DIM);
  float4 v[8];
#pragma unroll
  for (int u = 0; u < 8; ++u) v[u] = s4[u * 32 + lane];
  unsigned taken; topk_select(v, lane, row, zval, zidx, taken);
  (void)taken;
}

// c = b + sum_t val_t * ST[idx_t]; z = hard_thr(c). 1 wave per row.
// final_flag: also write dense z to zdense and accumulate density.
__global__ __launch_bounds__(256)
void sparse_step(const float* __restrict__ b, const float* __restrict__ STm,
                 float* __restrict__ zval, int* __restrict__ zidx,
                 int final_flag, float* __restrict__ zdense,
                 float* __restrict__ dens)
{
  const int lane = threadIdx.x & 31;
  const int row  = blockIdx.x * 8 + (threadIdx.x >> 5);
  float vals[OMEGA]; int idxs[OMEGA];
#pragma unroll
  for (int t = 0; t < OMEGA; ++t) {
    vals[t] = zval[row * OMEGA + t];
    idxs[t] = zidx[row * OMEGA + t];
  }
  const float4* b4 = (const float4*)(b + (size_t)row * D_DIM);
  float4 v[8];
#pragma unroll
  for (int u = 0; u < 8; ++u) v[u] = b4[u * 32 + lane];
#pragma unroll 1
  for (int t = 0; t < OMEGA; ++t) {
    const float4* st4 = (const float4*)(STm + (size_t)idxs[t] * D_DIM);
    const float a = vals[t];
#pragma unroll
    for (int u = 0; u < 8; ++u) {
      float4 w = st4[u * 32 + lane];
      v[u].x = fmaf(a, w.x, v[u].x);
      v[u].y = fmaf(a, w.y, v[u].y);
      v[u].z = fmaf(a, w.z, v[u].z);
      v[u].w = fmaf(a, w.w, v[u].w);
    }
  }
  unsigned taken; topk_select(v, lane, row, zval, zidx, taken);
  if (final_flag) {
    float4* o4 = (float4*)(zdense + (size_t)row * D_DIM);
#pragma unroll
    for (int u = 0; u < 8; ++u) {
      const int jb = (u * 32 + lane) << 2;
      float4 o;
      o.x = ((taken >> (u*4+0)) & 1u) ? v[u].x : 0.0f;
      o.y = ((taken >> (u*4+1)) & 1u) ? v[u].y : 0.0f;
      o.z = ((taken >> (u*4+2)) & 1u) ? v[u].z : 0.0f;
      o.w = ((taken >> (u*4+3)) & 1u) ? v[u].w : 0.0f;
      o4[u * 32 + lane] = o;
      if ((taken >> (u*4+0)) & 1u) atomicAdd(&dens[(jb+0) & 511], o.x * o.x);
      if ((taken >> (u*4+1)) & 1u) atomicAdd(&dens[(jb+1) & 511], o.y * o.y);
      if ((taken >> (u*4+2)) & 1u) atomicAdd(&dens[(jb+2) & 511], o.z * o.z);
      if ((taken >> (u*4+3)) & 1u) atomicAdd(&dens[(jb+3) & 511], o.w * o.w);
    }
  }
}

// spectral density from sparse list
__global__ __launch_bounds__(256)
void dens_from_list(const float* __restrict__ zval, const int* __restrict__ zidx,
                    float* __restrict__ dens, int n)
{
  int g = blockIdx.x * blockDim.x + threadIdx.x;
  if (g < n) { float v = zval[g]; atomicAdd(&dens[zidx[g] & 511], v * v); }
}

// (x - min) / (max - min + eps) over 512 entries, single block
__global__ __launch_bounds__(512)
void minmax_norm(const float* __restrict__ in, float* __restrict__ out)
{
  __shared__ float smn[512];
  __shared__ float smx[512];
  const int tid = threadIdx.x;
  const float v = in[tid];
  smn[tid] = v; smx[tid] = v;
  __syncthreads();
  for (int s = 256; s > 0; s >>= 1) {
    if (tid < s) {
      smn[tid] = fminf(smn[tid], smn[tid + s]);
      smx[tid] = fmaxf(smx[tid], smx[tid + s]);
    }
    __syncthreads();
  }
  out[tid] = (v - smn[0]) / (smx[0] - smn[0] + 1e-8f);
}

// density of previous windows: mdprev[p][w] += sum_i pw[p,i,w]^2 + pw[p,i,w+512]^2
__global__ __launch_bounds__(256)
void prev_density(const float* __restrict__ pw, float* __restrict__ mdprev)
{
  const int p  = blockIdx.y;
  const int r0 = blockIdx.x * 256;
  const int tid = threadIdx.x;
  const float* base = pw + (size_t)p * BATCH * D_DIM;
  float a0 = 0.f, a1 = 0.f;
  for (int i = 0; i < 256; ++i) {
    const float* rowp = base + (size_t)(r0 + i) * D_DIM;
    float x0 = rowp[tid],       y0 = rowp[tid + 512];
    float x1 = rowp[tid + 256], y1 = rowp[tid + 768];
    a0 = fmaf(x0, x0, fmaf(y0, y0, a0));
    a1 = fmaf(x1, x1, fmaf(y1, y1, a1));
  }
  atomicAdd(&mdprev[p * 512 + tid],       a0);
  atomicAdd(&mdprev[p * 512 + tid + 256], a1);
}

// per-p minmax-normalize, dot with mDn, softmax over P -> att[8]
__global__ __launch_bounds__(512)
void attention_k(const float* __restrict__ mdprev, const float* __restrict__ mdn,
                 float* __restrict__ att)
{
  __shared__ float sh[512];
  __shared__ float araw[P_PREV];
  const int tid = threadIdx.x;
  const float dn = mdn[tid];
  for (int p = 0; p < P_PREV; ++p) {
    const float v = mdprev[p * 512 + tid];
    sh[tid] = v; __syncthreads();
    for (int s = 256; s > 0; s >>= 1) { if (tid < s) sh[tid] = fminf(sh[tid], sh[tid + s]); __syncthreads(); }
    const float mn = sh[0]; __syncthreads();
    sh[tid] = v; __syncthreads();
    for (int s = 256; s > 0; s >>= 1) { if (tid < s) sh[tid] = fmaxf(sh[tid], sh[tid + s]); __syncthreads(); }
    const float mx = sh[0]; __syncthreads();
    sh[tid] = ((v - mn) / (mx - mn + 1e-8f)) * dn; __syncthreads();
    for (int s = 256; s > 0; s >>= 1) { if (tid < s) sh[tid] += sh[tid + s]; __syncthreads(); }
    if (tid == 0) araw[p] = sh[0];
    __syncthreads();
  }
  if (tid == 0) {
    const float inv = 0.04419417382415922f;   // 1/sqrt(512)
    float m = -1e30f;
    for (int p = 0; p < P_PREV; ++p) m = fmaxf(m, araw[p] * inv);
    float s = 0.f, e[P_PREV];
    for (int p = 0; p < P_PREV; ++p) { e[p] = expf(araw[p] * inv - m); s += e[p]; }
    for (int p = 0; p < P_PREV; ++p) att[p] = e[p] / s;
  }
}

// z_att = sum_p clip(pw[p], +-150) * att[p] * lambda2  (float4 per thread)
__global__ __launch_bounds__(256)
void zatt_k(const float* __restrict__ pw, const float* __restrict__ att,
            const float* __restrict__ lambda2p, float* __restrict__ out)
{
  const size_t g = (size_t)blockIdx.x * blockDim.x + threadIdx.x;  // float4 idx
  const float l2 = lambda2p[0];
  float4 s = {0.f, 0.f, 0.f, 0.f};
#pragma unroll
  for (int p = 0; p < P_PREV; ++p) {
    const float a = att[p];
    float4 v = ((const float4*)pw)[(size_t)p * ((size_t)BATCH * D_DIM / 4) + g];
    v.x = fminf(fmaxf(v.x, -150.f), 150.f);
    v.y = fminf(fmaxf(v.y, -150.f), 150.f);
    v.z = fminf(fmaxf(v.z, -150.f), 150.f);
    v.w = fminf(fmaxf(v.w, -150.f), 150.f);
    s.x = fmaf(v.x, a, s.x); s.y = fmaf(v.y, a, s.y);
    s.z = fmaf(v.z, a, s.z); s.w = fmaf(v.w, a, s.w);
  }
  s.x *= l2; s.y *= l2; s.z *= l2; s.w *= l2;
  ((float4*)out)[g] = s;
}

// ST = S0^T (LDS-tiled)
__global__ __launch_bounds__(256)
void transpose_k(const float* __restrict__ S, float* __restrict__ STm)
{
  __shared__ float t[32][33];
  const int bx = blockIdx.x * 32, by = blockIdx.y * 32;
  const int tx = threadIdx.x & 31, ty = threadIdx.x >> 5;  // 32x8
#pragma unroll
  for (int i = 0; i < 32; i += 8)
    t[ty + i][tx] = S[(size_t)(by + ty + i) * D_DIM + bx + tx];
  __syncthreads();
#pragma unroll
  for (int i = 0; i < 32; i += 8)
    STm[(size_t)(bx + ty + i) * D_DIM + by + tx] = t[tx][ty + i];
}

__global__ void zero_k(float* __restrict__ p, int n)
{
  int g = blockIdx.x * blockDim.x + threadIdx.x;
  if (g < n) p[g] = 0.f;
}

// =====================================================================
extern "C" void kernel_launch(void* const* d_in, const int* in_sizes, int n_in,
                              void* d_out, int out_size, void* d_ws, size_t ws_size,
                              hipStream_t stream)
{
  (void)in_sizes; (void)n_in; (void)out_size; (void)ws_size;
  const float* x    = (const float*)d_in[0];
  const float* pw   = (const float*)d_in[1];
  const float* Wd   = (const float*)d_in[2];
  const float* S    = (const float*)d_in[3];
  const float* lam2 = (const float*)d_in[4];
  const float* Lp   = (const float*)d_in[5];
  float* out = (float*)d_out;
  float* ws  = (float*)d_ws;

  float*  b      = ws + OFS_B;
  float*  STm    = ws + OFS_STR;
  __bf16* Ahp    = (__bf16*)(ws + OFS_AH);
  __bf16* Alp    = (__bf16*)(ws + OFS_AL);
  __bf16* Whp    = (__bf16*)(ws + OFS_WH);
  __bf16* Wlp    = (__bf16*)(ws + OFS_WL);
  __bf16* Shp    = (__bf16*)(ws + OFS_SH);
  __bf16* Slp    = (__bf16*)(ws + OFS_SL);
  float*  zval   = ws + OFS_ZVAL;
  int*    zidx   = (int*)(ws + OFS_ZIDX);
  float*  dens0  = ws + OFS_DENS0;
  float*  dens1  = ws + OFS_DENS1;
  float*  mdprev = ws + OFS_MDPREV;
  float*  mdn    = ws + OFS_MDN;
  float*  att    = ws + OFS_ATT;
  float*  zout   = out + W_HALF;   // z_att -> c1 -> final dense z (in output)
  float*  mdout  = out;            // mD_normalized [512]

  const dim3 gemm_grid(D_DIM / BN, BATCH / BM);
  const int n4_big   = BATCH * D_DIM / 4;   // 1048576
  const int n4_small = D_DIM * D_DIM / 4;   // 262144

  // zero the atomic accumulators (dens0, dens1, mdprev are contiguous: 5120)
  zero_k<<<(5120 + 255) / 256, 256, 0, stream>>>(dens0, 5120);
  transpose_k<<<dim3(32, 32), 256, 0, stream>>>(S, STm);

  // pre-pack hi/lo bf16 once per matrix
  prepack_bf16x2<<<n4_big / 256, 256, 0, stream>>>(x, Ahp, Alp, n4_big);
  prepack_bf16x2<<<n4_small / 256, 256, 0, stream>>>(Wd, Whp, Wlp, n4_small);
  prepack_bf16x2<<<n4_small / 256, 256, 0, stream>>>(S, Shp, Slp, n4_small);

  // b = (x @ Wd^T) / L
  gemm_nt_bf16pk<<<gemm_grid, 256, 0, stream>>>(Ahp, Alp, Whp, Wlp, b,
                                                nullptr, Lp,
                                                BATCH, D_DIM, D_DIM);
  // warm-up: z = thr(b); c = b + z@S^T; z = thr(c)
  hardthr_rows<<<BATCH / 8, 256, 0, stream>>>(b, zval, zidx);
  sparse_step<<<BATCH / 8, 256, 0, stream>>>(b, STm, zval, zidx, 0, nullptr, nullptr);

  // spectral density of warm-up z, normalized
  dens_from_list<<<(BATCH * OMEGA + 255) / 256, 256, 0, stream>>>(
      zval, zidx, dens0, BATCH * OMEGA);
  minmax_norm<<<1, 512, 0, stream>>>(dens0, mdn);

  // attention over previous windows
  prev_density<<<dim3(16, P_PREV), 256, 0, stream>>>(pw, mdprev);
  attention_k<<<1, 512, 0, stream>>>(mdprev, mdn, att);
  zatt_k<<<n4_big / 256, 256, 0, stream>>>(pw, att, lam2, zout);

  // iteration 1 (dense z): c1 = b + z_att @ S^T ; z = thr(c1)
  prepack_bf16x2<<<n4_big / 256, 256, 0, stream>>>(zout, Ahp, Alp, n4_big);
  gemm_nt_bf16pk<<<gemm_grid, 256, 0, stream>>>(Ahp, Alp, Shp, Slp, zout,
                                                b, nullptr,
                                                BATCH, D_DIM, D_DIM);
  hardthr_rows<<<BATCH / 8, 256, 0, stream>>>(zout, zval, zidx);

  // iterations 2..10: z is OMEGA-sparse -> rank-10 row updates
  for (int it = 0; it < N_IT - 1; ++it) {
    const int fin = (it == N_IT - 2);
    sparse_step<<<BATCH / 8, 256, 0, stream>>>(b, STm, zval, zidx, fin,
                                               zout, dens1);
  }
  // final mD_normalized -> d_out[0:512]
  minmax_norm<<<1, 512, 0, stream>>>(dens1, mdout);
}
